// FULL_BNN_1W1A_47090021433798
// MI455X (gfx1250) — 8-bit matrix operand order repaired and checked by reading, compiled, not run on hardware
//
#include <hip/hip_runtime.h>
#include <stdint.h>

typedef __attribute__((ext_vector_type(8))) int v8i;
typedef __attribute__((ext_vector_type(4))) int v4i;
typedef __attribute__((ext_vector_type(2))) int v2i;

#define BATCH 256

__global__ void k_binw(const float* __restrict__ w, int8_t* __restrict__ wb,
                       int Co, int Ci) {
  int idx = blockIdx.x * 256 + threadIdx.x;
  int total = Co * 9 * Ci;
  if (idx >= total) return;
  int ci = idx % Ci;
  int t  = (idx / Ci) % 9;
  int co = idx / (9 * Ci);
  int kh = t / 3, kw = t % 3;
  float v = w[((co * Ci + ci) * 3 + kh) * 3 + kw];
  wb[idx] = (v >= 0.0f) ? (int8_t)1 : (int8_t)-1;
}

__global__ void k_pack_input(const float* __restrict__ x, int8_t* __restrict__ act) {
  int idx = blockIdx.x * 256 + threadIdx.x;
  if (idx >= BATCH * 3 * 32 * 32) return;
  int b = idx / (3 * 1024);
  int r = idx % (3 * 1024);
  int c = r / 1024;
  int y = (r >> 5) & 31;
  int xx = r & 31;
  float v = x[idx];
  act[(((size_t)b * 34 + (y + 1)) * 34 + (xx + 1)) * 3 + c] = (v >= 0.0f) ? 1 : -1;
}

__global__ void k_conv0(const int8_t* __restrict__ act, const int8_t* __restrict__ wb,
                        int16_t* __restrict__ out) {
  int idx = blockIdx.x * 256 + threadIdx.x;
  if (idx >= BATCH * 32 * 32 * 128) return;
  int co = idx & 127;
  int p  = idx >> 7;
  int xx = p & 31;
  int y  = (p >> 5) & 31;
  int b  = p >> 10;
  int acc = 0;
#pragma unroll
  for (int kh = 0; kh < 3; ++kh)
#pragma unroll
    for (int kw = 0; kw < 3; ++kw) {
      const int8_t* ap = act + (((size_t)b * 34 + (y + kh)) * 34 + (xx + kw)) * 3;
      const int8_t* wp = wb + co * 27 + (kh * 3 + kw) * 3;
      acc += (int)ap[0] * (int)wp[0] + (int)ap[1] * (int)wp[1] + (int)ap[2] * (int)wp[2];
    }
  out[idx] = (int16_t)acc;
}

template <int Ci, int Co>
__global__ void k_conv_wmma(const int8_t* __restrict__ act, const int8_t* __restrict__ wb,
                            int16_t* __restrict__ out, int logW, int logH) {
  const int lane = threadIdx.x & 31;
  const int col  = lane & 15;
  const int half = lane >> 4;
  const int co0  = blockIdx.y << 5;

  const unsigned waveId = (blockIdx.x << 3) | (threadIdx.x >> 5);
  const unsigned q0 = (waveId << 5) | col;
  const unsigned q1 = q0 | 16;
  const int W  = 1 << logW;
  const int H  = 1 << logH;
  const int x0 = q0 & (W - 1), y0 = (q0 >> logW) & (H - 1), b0 = q0 >> (logW + logH);
  const int x1 = q1 & (W - 1), y1 = (q1 >> logW) & (H - 1), b1 = q1 >> (logW + logH);
  const int Wp = W + 2;
  constexpr int KC = Ci >> 6;

  const int8_t* __restrict__ arow0 = wb + (size_t)(co0 + col) * 9 * Ci;
  const int8_t* __restrict__ arow1 = arow0 + (size_t)16 * 9 * Ci;
  v8i acc00 = (v8i)0, acc01 = (v8i)0, acc10 = (v8i)0, acc11 = (v8i)0;

#pragma unroll
  for (int tap = 0; tap < 9; ++tap) {
    const int kh = tap / 3, kw = tap - kh * 3;
    const int8_t* bbase0 = act + (((size_t)b0 * (H + 2) + (y0 + kh)) * Wp + (x0 + kw)) * Ci;
    const int8_t* bbase1 = act + (((size_t)b1 * (H + 2) + (y1 + kh)) * Wp + (x1 + kw)) * Ci;
    const int8_t* abase0 = arow0 + tap * Ci;
    const int8_t* abase1 = arow1 + tap * Ci;
    if (tap < 8) {
      const int nkh = (tap + 1) / 3, nkw = (tap + 1) - nkh * 3;
      __builtin_prefetch(act + (((size_t)b0 * (H + 2) + (y0 + nkh)) * Wp + (x0 + nkw)) * Ci, 0, 0);
      __builtin_prefetch(act + (((size_t)b1 * (H + 2) + (y1 + nkh)) * Wp + (x1 + nkw)) * Ci, 0, 0);
    }
#pragma unroll
    for (int cc = 0; cc < KC; ++cc) {
      const int ci0 = cc << 6;
      v4i p0 = *(const v4i*)(bbase0 + ci0 + half * 16);
      v4i p1 = *(const v4i*)(bbase0 + ci0 + 32 + half * 16);
      v4i p2 = *(const v4i*)(bbase1 + ci0 + half * 16);
      v4i p3 = *(const v4i*)(bbase1 + ci0 + 32 + half * 16);
      const int8_t* ap0 = abase0 + ci0 + half * 16;
      const int8_t* ap1 = abase1 + ci0 + half * 16;
      v2i a00 = *(const v2i*)(ap0);      v2i a01 = *(const v2i*)(ap0 + 8);
      v2i a02 = *(const v2i*)(ap0 + 32); v2i a03 = *(const v2i*)(ap0 + 40);
      v2i a10 = *(const v2i*)(ap1);      v2i a11 = *(const v2i*)(ap1 + 8);
      v2i a12 = *(const v2i*)(ap1 + 32); v2i a13 = *(const v2i*)(ap1 + 40);
      v8i A0, A1, B0, B1;
      A0[0] = a00[0]; A0[1] = a00[1]; A0[2] = a01[0]; A0[3] = a01[1];
      A0[4] = a02[0]; A0[5] = a02[1]; A0[6] = a03[0]; A0[7] = a03[1];
      A1[0] = a10[0]; A1[1] = a10[1]; A1[2] = a11[0]; A1[3] = a11[1];
      A1[4] = a12[0]; A1[5] = a12[1]; A1[6] = a13[0]; A1[7] = a13[1];
      B0[0] = p0[0]; B0[1] = p0[1]; B0[2] = p0[2]; B0[3] = p0[3];
      B0[4] = p1[0]; B0[5] = p1[1]; B0[6] = p1[2]; B0[7] = p1[3];
      B1[0] = p2[0]; B1[1] = p2[1]; B1[2] = p2[2]; B1[3] = p2[3];
      B1[4] = p3[0]; B1[5] = p3[1]; B1[6] = p3[2]; B1[7] = p3[3];
      acc00 = __builtin_amdgcn_wmma_i32_16x16x64_iu8(true, A0, true, B0, acc00, false, false);
      acc01 = __builtin_amdgcn_wmma_i32_16x16x64_iu8(true, A0, true, B1, acc01, false, false);
      acc10 = __builtin_amdgcn_wmma_i32_16x16x64_iu8(true, A1, true, B0, acc10, false, false);
      acc11 = __builtin_amdgcn_wmma_i32_16x16x64_iu8(true, A1, true, B1, acc11, false, false);
    }
  }

  auto pack = [](const v8i& a) {
    int4 st;
    st.x = (a[0] & 0xffff) | (a[1] << 16);
    st.y = (a[2] & 0xffff) | (a[3] << 16);
    st.z = (a[4] & 0xffff) | (a[5] << 16);
    st.w = (a[6] & 0xffff) | (a[7] << 16);
    return st;
  };
  int16_t* o0 = out + (size_t)q0 * Co + co0 + half * 8;
  int16_t* o1 = out + (size_t)q1 * Co + co0 + half * 8;
  *(int4*)(o0)      = pack(acc00);
  *(int4*)(o0 + 16) = pack(acc10);
  *(int4*)(o1)      = pack(acc01);
  *(int4*)(o1 + 16) = pack(acc11);
}

__global__ void k_stats(const int16_t* __restrict__ conv, long long* __restrict__ stats,
                        int P, int logCo) {
  int c = threadIdx.x;
  long long s = 0, sq = 0;
  for (int p = blockIdx.x; p < P; p += gridDim.x) {
    int v = (int)conv[((size_t)p << logCo) + c];
    s += v;
    sq += (long long)(v * v);
  }
  atomicAdd((unsigned long long*)&stats[2 * c],     (unsigned long long)s);
  atomicAdd((unsigned long long*)&stats[2 * c + 1], (unsigned long long)sq);
}

__global__ void k_coeff(const long long* __restrict__ stats, const float* __restrict__ gamma,
                        const float* __restrict__ beta, float2* __restrict__ coeff,
                        int Co, float invN) {
  int c = blockIdx.x * blockDim.x + threadIdx.x;
  if (c >= Co) return;
  float mean = (float)stats[2 * c] * invN;
  float var  = (float)stats[2 * c + 1] * invN - mean * mean;
  float a = gamma[c] * rsqrtf(var + 1e-5f);
  float2 r; r.x = a; r.y = beta[c] - a * mean;
  coeff[c] = r;
}

__global__ void k_bnsignpool(const int16_t* __restrict__ conv, const float2* __restrict__ coeff,
                             int8_t* __restrict__ actout, int logCo, int pool,
                             int logWo, int logHo) {
  const int Co = 1 << logCo;
  const int Wo = 1 << logWo, Ho = 1 << logHo;
  const int W = Wo << pool, H = Ho << pool;
  long idx = (long)blockIdx.x * 256 + threadIdx.x;
  long total = (long)BATCH << (logCo + logWo + logHo);
  if (idx >= total) return;
  const int c  = (int)(idx & (Co - 1));
  const long p = idx >> logCo;
  const int xo = (int)(p & (Wo - 1));
  const int yo = (int)((p >> logWo) & (Ho - 1));
  const int b  = (int)(p >> (logWo + logHo));
  const float2 ac = coeff[c];
  int8_t r = -1;
  const int y0 = yo << pool, x0 = xo << pool, n = 1 << pool;
  for (int dy = 0; dy < n; ++dy)
    for (int dx = 0; dx < n; ++dx) {
      int v = (int)conv[((((size_t)b * H + (y0 + dy)) * W + (x0 + dx)) << logCo) + c];
      float f = ac.x * (float)v + ac.y;
      if (f >= 0.0f) r = 1;
    }
  actout[(((size_t)b * (Ho + 2) + (yo + 1)) * (Wo + 2) + (xo + 1)) * Co + c] = r;
}

__global__ void k_fc(const int8_t* __restrict__ act, const float* __restrict__ fcw,
                     const float* __restrict__ fcb, float* __restrict__ out) {
  int b = blockIdx.x;
  int t = threadIdx.x;
  __shared__ float sh[256];
  for (int o = 0; o < 10; ++o) {
    float s = 0.0f;
    for (int i = t; i < 2048; i += 256) {
      int c = i >> 4, y = (i >> 2) & 3, xx = i & 3;
      int8_t a = act[(((size_t)b * 6 + (y + 1)) * 6 + (xx + 1)) * 128 + c];
      s += (float)a * fcw[o * 2048 + i];
    }
    sh[t] = s;
    __syncthreads();
    for (int off = 128; off > 0; off >>= 1) {
      if (t < off) sh[t] += sh[t + off];
      __syncthreads();
    }
    if (t == 0) out[b * 10 + o] = sh[0] + fcb[o];
    __syncthreads();
  }
}

extern "C" void kernel_launch(void* const* d_in, const int* in_sizes, int n_in,
                              void* d_out, int out_size, void* d_ws, size_t ws_size,
                              hipStream_t stream) {
  const float* x = (const float*)d_in[0];
  const float *w[6], *g[6], *bt[6];
  for (int i = 0; i < 6; ++i) {
    w[i]  = (const float*)d_in[1 + 3 * i];
    g[i]  = (const float*)d_in[2 + 3 * i];
    bt[i] = (const float*)d_in[3 + 3 * i];
  }
  const float* fcw = (const float*)d_in[19];
  const float* fcb = (const float*)d_in[20];
  float* out = (float*)d_out;

  char* ws = (char*)d_ws;
  size_t off = 0;
  auto alloc = [&](size_t bytes) -> void* {
    void* p = ws + off;
    off = (off + bytes + 255) & ~(size_t)255;
    return p;
  };
  static const int CiT[6] = {3, 128, 256, 256, 256, 128};
  static const int CoT[6] = {128, 256, 256, 256, 128, 128};
  int8_t* wb[6];
  for (int i = 0; i < 6; ++i) wb[i] = (int8_t*)alloc((size_t)CoT[i] * 9 * CiT[i]);
  int8_t*  act0  = (int8_t*) alloc((size_t)BATCH * 34 * 34 * 3);
  int8_t*  actA  = (int8_t*) alloc((size_t)BATCH * 34 * 34 * 128);
  int8_t*  actB  = (int8_t*) alloc((size_t)BATCH * 34 * 34 * 128);
  int16_t* conv  = (int16_t*)alloc((size_t)BATCH * 32 * 32 * 256 * 2);
  long long* stats = (long long*)alloc(2 * 256 * sizeof(long long));
  float2*  coeff = (float2*) alloc(256 * sizeof(float2));

  for (int i = 0; i < 6; ++i) {
    int total = CoT[i] * 9 * CiT[i];
    k_binw<<<(total + 255) / 256, 256, 0, stream>>>(w[i], wb[i], CoT[i], CiT[i]);
  }

  hipMemsetAsync(act0, 0xFF, (size_t)BATCH * 34 * 34 * 3, stream);
  k_pack_input<<<(BATCH * 3 * 32 * 32 + 255) / 256, 256, 0, stream>>>(x, act0);
  k_conv0<<<(BATCH * 32 * 32 * 128 + 255) / 256, 256, 0, stream>>>(act0, wb[0], conv);
  {
    int P = BATCH * 32 * 32;
    hipMemsetAsync(stats, 0, 2 * 128 * sizeof(long long), stream);
    k_stats<<<1024, 128, 0, stream>>>(conv, stats, P, 7);
    k_coeff<<<1, 256, 0, stream>>>(stats, g[0], bt[0], coeff, 128, 1.0f / (float)P);
    hipMemsetAsync(actA, 0xFF, (size_t)BATCH * 34 * 34 * 128, stream);
    long total = (long)BATCH * 32 * 32 * 128;
    k_bnsignpool<<<(int)((total + 255) / 256), 256, 0, stream>>>(conv, coeff, actA,
                                                                 7  , 0  ,
                                                                 5  , 5  );
  }

  struct Cfg { int logH, logW, Ci, Co, pool; };
  static const Cfg cfg[5] = {
    {5, 5, 128, 256, 1},
    {4, 4, 256, 256, 0},
    {4, 4, 256, 256, 1},
    {3, 3, 256, 128, 0},
    {3, 3, 128, 128, 1},
  };
  for (int i = 0; i < 5; ++i) {
    const Cfg& c = cfg[i];
    int8_t* in  = (i & 1) ? actB : actA;
    int8_t* oub = (i & 1) ? actA : actB;
    const int H = 1 << c.logH, W = 1 << c.logW;
    const long P = (long)BATCH * H * W;
    dim3 grid((unsigned)(P / 32 / 8), (unsigned)(c.Co / 32));

    if (c.Ci == 128 && c.Co == 256)
      k_conv_wmma<128, 256><<<grid, 256, 0, stream>>>(in, wb[i + 1], conv, c.logW, c.logH);
    else if (c.Ci == 256 && c.Co == 256)
      k_conv_wmma<256, 256><<<grid, 256, 0, stream>>>(in, wb[i + 1], conv, c.logW, c.logH);
    else if (c.Ci == 256 && c.Co == 128)
      k_conv_wmma<256, 128><<<grid, 256, 0, stream>>>(in, wb[i + 1], conv, c.logW, c.logH);
    else
      k_conv_wmma<128, 128><<<grid, 256, 0, stream>>>(in, wb[i + 1], conv, c.logW, c.logH);

    const int logCo = (c.Co == 256) ? 8 : 7;
    hipMemsetAsync(stats, 0, 2 * c.Co * sizeof(long long), stream);
    k_stats<<<1024, c.Co, 0, stream>>>(conv, stats, (int)P, logCo);
    k_coeff<<<1, 256, 0, stream>>>(stats, g[i + 1], bt[i + 1], coeff, c.Co, 1.0f / (float)P);

    const int logHo = c.logH - c.pool, logWo = c.logW - c.pool;
    const int Ho = 1 << logHo, Wo = 1 << logWo;
    hipMemsetAsync(oub, 0xFF, (size_t)BATCH * (Ho + 2) * (Wo + 2) * c.Co, stream);
    const long total = (long)BATCH << (logHo + logWo + logCo);
    k_bnsignpool<<<(int)((total + 255) / 256), 256, 0, stream>>>(conv, coeff, oub,
                                                                 logCo, c.pool, logWo, logHo);
  }

  k_fc<<<BATCH, 256, 0, stream>>>(actB, fcw, fcb, out);
}
